// GPTSelfAttention_49649821941897
// MI455X (gfx1250) — matrix operand order repaired and checked by reading, not compiled, not run on hardware
//
#include <hip/hip_runtime.h>

// ---------------- Types ----------------
typedef __bf16 bf16;
typedef __attribute__((ext_vector_type(16))) __bf16 bf16x16;
typedef __attribute__((ext_vector_type(8)))  float  f32x8;

#define WMMA_BF16(a, b, c) \
  __builtin_amdgcn_wmma_f32_16x16x32_bf16(false, (a), false, (b), (short)0, (c), false, false)

// Problem constants
#define Bq   2
#define Sq   2048
#define Dq   1024
#define Hq   16
#define HDq  64
#define N3D  3072
#define MROWS (Bq*Sq)          // 4096
#define SROW 2052              // padded LDS row stride (floats), keeps 16B alignment

// Pack WMMA A-operand (16 bf16) from one contiguous 16-float chunk (fp32 source)
__device__ __forceinline__ bf16x16 packA_f32(const float* ap) {
  float4 f0 = *(const float4*)(ap);
  float4 f1 = *(const float4*)(ap + 4);
  float4 f2 = *(const float4*)(ap + 8);
  float4 f3 = *(const float4*)(ap + 12);
  bf16x16 a;
  a[0]=(bf16)f0.x; a[1]=(bf16)f0.y; a[2]=(bf16)f0.z; a[3]=(bf16)f0.w;
  a[4]=(bf16)f1.x; a[5]=(bf16)f1.y; a[6]=(bf16)f1.z; a[7]=(bf16)f1.w;
  a[8]=(bf16)f2.x; a[9]=(bf16)f2.y; a[10]=(bf16)f2.z; a[11]=(bf16)f2.w;
  a[12]=(bf16)f3.x; a[13]=(bf16)f3.y; a[14]=(bf16)f3.z; a[15]=(bf16)f3.w;
  return a;
}
// Pack WMMA A-operand from one contiguous 16-bf16 chunk (bf16 source)
__device__ __forceinline__ bf16x16 packA_bf16(const bf16* ap) {
  bf16x16 a;
#pragma unroll
  for (int i = 0; i < 8; ++i) { a[i] = ap[i]; a[8 + i] = ap[8 + i]; }
  return a;
}

// ---------------- Kernel 1: fp32 -> bf16 transpose of weights ----------------
// Wt[n*1024 + k] = (bf16) W[k*N + n], K fixed at 1024
__global__ void cvt_transpose_k(const float* __restrict__ W, bf16* __restrict__ Wt, int N) {
  long idx = (long)blockIdx.x * blockDim.x + threadIdx.x;   // over N*1024
  int n = (int)(idx >> 10);
  int k = (int)(idx & 1023);
  Wt[idx] = (bf16)W[(size_t)k * N + n];
}

// ---------------- Kernel 2: QKV GEMM  C = x @ Wqkv + b  (bf16 WMMA, pipelined) ----------------
// grid (N/64=48, M/64=64), block 128 (4 waves). Wave w owns rows Mb+16w..+15, all 64 cols.
__global__ void gemm_qkv_k(const float* __restrict__ x, const bf16* __restrict__ Wt,
                           const float* __restrict__ bias,
                           bf16* __restrict__ Q, bf16* __restrict__ Km, bf16* __restrict__ Vt) {
  const int tid  = threadIdx.x;
  const int wave = tid >> 5, lane = tid & 31;
  const int half = lane >> 4, ln = lane & 15;
  const int Mb = blockIdx.y * 64 + wave * 16;
  const int Nb = blockIdx.x * 64;

  f32x8 acc0 = {}, acc1 = {}, acc2 = {}, acc3 = {};
  const float* xrow = x + (size_t)(Mb + ln) * Dq + half * 16;
  const bf16*  brow = Wt + (size_t)(Nb + ln) * Dq + half * 16;

  // stage 0 preload
  bf16x16 a  = packA_f32(xrow);
  bf16x16 b0 = *(const bf16x16*)(brow);
  bf16x16 b1 = *(const bf16x16*)(brow + 16 * Dq);
  bf16x16 b2 = *(const bf16x16*)(brow + 32 * Dq);
  bf16x16 b3 = *(const bf16x16*)(brow + 48 * Dq);

#pragma unroll 4
  for (int ks = 32; ks < Dq; ks += 32) {
    // prefetch next k-step operands while current WMMAs execute
    bf16x16 an  = packA_f32(xrow + ks);
    const bf16* bp = brow + ks;
    __builtin_prefetch(bp + 32, 0, 1);
    bf16x16 n0 = *(const bf16x16*)(bp);
    bf16x16 n1 = *(const bf16x16*)(bp + 16 * Dq);
    bf16x16 n2 = *(const bf16x16*)(bp + 32 * Dq);
    bf16x16 n3 = *(const bf16x16*)(bp + 48 * Dq);
    acc0 = WMMA_BF16(a, b0, acc0);
    acc1 = WMMA_BF16(a, b1, acc1);
    acc2 = WMMA_BF16(a, b2, acc2);
    acc3 = WMMA_BF16(a, b3, acc3);
    a = an; b0 = n0; b1 = n1; b2 = n2; b3 = n3;
  }
  acc0 = WMMA_BF16(a, b0, acc0);
  acc1 = WMMA_BF16(a, b1, acc1);
  acc2 = WMMA_BF16(a, b2, acc2);
  acc3 = WMMA_BF16(a, b3, acc3);

#pragma unroll
  for (int t = 0; t < 4; ++t) {
    const f32x8 acc = (t == 0) ? acc0 : (t == 1) ? acc1 : (t == 2) ? acc2 : acc3;
    const int colg = Nb + t * 16 + ln;          // [0,3072)
    const float bv = bias[colg];
    const int region = colg >> 10;              // 0=Q 1=K 2=V
    const int d = colg & 1023;
    const int h = d >> 6, hd = d & 63;
#pragma unroll
    for (int v = 0; v < 8; ++v) {
      const int rg = Mb + v + half * 8;         // global row [0,4096)
      const int b = rg >> 11, s = rg & 2047;
      const float val = acc[v] + bv;
      const size_t bh = (size_t)(b * Hq + h);
      if (region == 0)      Q [(bh * Sq + s) * HDq + hd] = (bf16)val;
      else if (region == 1) Km[(bh * Sq + s) * HDq + hd] = (bf16)val;
      else                  Vt[(bh * HDq + hd) * Sq + s] = (bf16)val;   // V transposed
    }
  }
}

// ---------------- Kernel 3: fused attention per 16-query tile ----------------
// grid = B*H*(S/16) = 4096, block 128 (4 waves). Dynamic LDS:
//   sc   [16][SROW]  fp32 score/prob rows (131 KB)
//   ctxp [4][16*64]  fp32 per-wave ctx partials (16 KB)
__global__ void attn_k(const bf16* __restrict__ Q, const bf16* __restrict__ Km,
                       const bf16* __restrict__ Vt,
                       float* __restrict__ probs, bf16* __restrict__ ctxb) {
  extern __shared__ float smem[];
  float* sc   = smem;                 // 16*SROW
  float* ctxp = smem + 16 * SROW;     // 4*1024

  const int tid  = threadIdx.x;
  const int wave = tid >> 5, lane = tid & 31;
  const int half = lane >> 4, ln = lane & 15;
  const int qt = blockIdx.x & 127;
  const int bh = blockIdx.x >> 7;                 // b*16+h
  const int nkt = qt + 1;                         // live 16-key tiles (causal)

  const bf16* Qbh = Q  + (size_t)bh * Sq * HDq;
  const bf16* Kbh = Km + (size_t)bh * Sq * HDq;
  const bf16* Vbh = Vt + (size_t)bh * HDq * Sq;
  float* probs_bh = probs + (size_t)bh * Sq * Sq;

  // ---- Phase 1: scores = Q Kᵀ * 1/8, causal mask, -> LDS (K tiles double-buffered) ----
  bf16x16 aq0, aq1;
  {
    const bf16* qp = Qbh + (size_t)(qt * 16 + ln) * HDq + half * 16;
#pragma unroll
    for (int i = 0; i < 8; ++i) {
      aq0[i] = qp[i];       aq0[8 + i] = qp[8 + i];
      aq1[i] = qp[32 + i];  aq1[8 + i] = qp[40 + i];
    }
  }
  {
    bf16x16 kb0, kb1;
    if (wave < nkt) {
      const bf16* kp = Kbh + (size_t)(wave * 16 + ln) * HDq + half * 16;
      kb0 = *(const bf16x16*)(kp);
      kb1 = *(const bf16x16*)(kp + 32);
    }
    for (int kt = wave; kt < nkt; kt += 4) {
      bf16x16 n0 = kb0, n1 = kb1;
      if (kt + 4 < nkt) {
        const bf16* kp = Kbh + (size_t)((kt + 4) * 16 + ln) * HDq + half * 16;
        n0 = *(const bf16x16*)(kp);
        n1 = *(const bf16x16*)(kp + 32);
      }
      f32x8 c = {};
      c = WMMA_BF16(aq0, kb0, c);
      c = WMMA_BF16(aq1, kb1, c);
      const int kcol = kt * 16 + ln;
#pragma unroll
      for (int v = 0; v < 8; ++v) {
        const int m = v + half * 8;
        float s = c[v] * 0.125f;                           // 1/sqrt(64)
        if (kt == qt && kcol > qt * 16 + m) s = -3.0e38f;  // causal
        sc[m * SROW + kcol] = s;
      }
      kb0 = n0; kb1 = n1;
    }
  }
  __syncthreads();

  // ---- Phase 2: softmax per row (8 threads/row, shfl-xor reductions) ----
  {
    const int row = tid >> 3, j = tid & 7;
    const int L = qt * 16 + row + 1;                     // valid length
    float* srow = sc + row * SROW;
    float mx = -3.4e38f;
    for (int i = j; i < L; i += 8) mx = fmaxf(mx, srow[i]);
#pragma unroll
    for (int o = 4; o >= 1; o >>= 1) mx = fmaxf(mx, __shfl_xor(mx, o, 8));
    float sum = 0.f;
    for (int i = j; i < L; i += 8) { float e = __expf(srow[i] - mx); srow[i] = e; sum += e; }
#pragma unroll
    for (int o = 4; o >= 1; o >>= 1) sum += __shfl_xor(sum, o, 8);
    const float inv = 1.0f / sum;
    float* prow = probs_bh + (size_t)(qt * 16 + row) * Sq;
    for (int i = j; i < L; i += 8) { float p = srow[i] * inv; srow[i] = p; prow[i] = p; }
    const int i0 = (L <= j) ? j : j + (((L - j + 7) >> 3) << 3);
    for (int i = i0; i < Sq; i += 8) { srow[i] = 0.f; prow[i] = 0.f; }   // causal tail = 0
  }
  __syncthreads();

  // ---- Phase 3: ctx = P @ V (bf16 WMMA, chunks split over waves, double-buffered) ----
  const int nch = (nkt + 1) >> 1;                        // 32-key chunks
  f32x8 c0 = {}, c1 = {}, c2 = {}, c3 = {};
  {
    const float* prow = sc + ln * SROW + half * 16;      // P operand base (LDS)
    const bf16*  vrow = Vbh + (size_t)ln * Sq + half * 16;
    bf16x16 pa, v0, v1, v2, v3;
    if (wave < nch) {
      const int kb = wave * 32;
      pa = packA_f32(prow + kb);                         // ds_load_b128 x4 + cvt
      v0 = *(const bf16x16*)(vrow + kb);
      v1 = *(const bf16x16*)(vrow + kb + 16 * Sq);
      v2 = *(const bf16x16*)(vrow + kb + 32 * Sq);
      v3 = *(const bf16x16*)(vrow + kb + 48 * Sq);
    }
    for (int c = wave; c < nch; c += 4) {
      bf16x16 pn = pa, n0 = v0, n1 = v1, n2 = v2, n3 = v3;
      if (c + 4 < nch) {
        const int kb = (c + 4) * 32;
        pn = packA_f32(prow + kb);
        n0 = *(const bf16x16*)(vrow + kb);
        n1 = *(const bf16x16*)(vrow + kb + 16 * Sq);
        n2 = *(const bf16x16*)(vrow + kb + 32 * Sq);
        n3 = *(const bf16x16*)(vrow + kb + 48 * Sq);
      }
      c0 = WMMA_BF16(pa, v0, c0);
      c1 = WMMA_BF16(pa, v1, c1);
      c2 = WMMA_BF16(pa, v2, c2);
      c3 = WMMA_BF16(pa, v3, c3);
      pa = pn; v0 = n0; v1 = n1; v2 = n2; v3 = n3;
    }
  }
#pragma unroll
  for (int v = 0; v < 8; ++v) {
    const int m = v + half * 8;
    float* cw = ctxp + wave * 1024 + m * 64 + ln;
    cw[0]  = c0[v];
    cw[16] = c1[v];
    cw[32] = c2[v];
    cw[48] = c3[v];
  }
  __syncthreads();

  // ---- Phase 4: reduce 4 wave partials, emit ctx (bf16, [B*S][D] layout) ----
  const int b = bh >> 4, h = bh & 15;
#pragma unroll
  for (int i = 0; i < 8; ++i) {
    const int e = tid * 8 + i;                           // [0,1024)
    const float s = ctxp[e] + ctxp[1024 + e] + ctxp[2048 + e] + ctxp[3072 + e];
    const int m = e >> 6, n = e & 63;
    ctxb[(size_t)(b * Sq + qt * 16 + m) * Dq + h * HDq + n] = (bf16)s;
  }
}

// ---------------- Kernel 4: output projection  out = ctx @ Wproj + b (pipelined) ----------------
__global__ void gemm_proj_k(const bf16* __restrict__ A, const bf16* __restrict__ Wt,
                            const float* __restrict__ bias, float* __restrict__ out) {
  const int tid  = threadIdx.x;
  const int wave = tid >> 5, lane = tid & 31;
  const int half = lane >> 4, ln = lane & 15;
  const int Mb = blockIdx.y * 64 + wave * 16;
  const int Nb = blockIdx.x * 64;

  f32x8 acc0 = {}, acc1 = {}, acc2 = {}, acc3 = {};
  const bf16* arow = A + (size_t)(Mb + ln) * Dq + half * 16;
  const bf16* brow = Wt + (size_t)(Nb + ln) * Dq + half * 16;

  bf16x16 a  = packA_bf16(arow);
  bf16x16 b0 = *(const bf16x16*)(brow);
  bf16x16 b1 = *(const bf16x16*)(brow + 16 * Dq);
  bf16x16 b2 = *(const bf16x16*)(brow + 32 * Dq);
  bf16x16 b3 = *(const bf16x16*)(brow + 48 * Dq);

#pragma unroll 4
  for (int ks = 32; ks < Dq; ks += 32) {
    bf16x16 an  = packA_bf16(arow + ks);
    const bf16* bp = brow + ks;
    __builtin_prefetch(bp + 32, 0, 1);
    bf16x16 n0 = *(const bf16x16*)(bp);
    bf16x16 n1 = *(const bf16x16*)(bp + 16 * Dq);
    bf16x16 n2 = *(const bf16x16*)(bp + 32 * Dq);
    bf16x16 n3 = *(const bf16x16*)(bp + 48 * Dq);
    acc0 = WMMA_BF16(a, b0, acc0);
    acc1 = WMMA_BF16(a, b1, acc1);
    acc2 = WMMA_BF16(a, b2, acc2);
    acc3 = WMMA_BF16(a, b3, acc3);
    a = an; b0 = n0; b1 = n1; b2 = n2; b3 = n3;
  }
  acc0 = WMMA_BF16(a, b0, acc0);
  acc1 = WMMA_BF16(a, b1, acc1);
  acc2 = WMMA_BF16(a, b2, acc2);
  acc3 = WMMA_BF16(a, b3, acc3);

#pragma unroll
  for (int t = 0; t < 4; ++t) {
    const f32x8 acc = (t == 0) ? acc0 : (t == 1) ? acc1 : (t == 2) ? acc2 : acc3;
    const int colg = Nb + t * 16 + ln;
    const float bv = bias[colg];
#pragma unroll
    for (int v = 0; v < 8; ++v) {
      const int rg = Mb + v + half * 8;
      out[(size_t)rg * Dq + colg] = acc[v] + bv;
    }
  }
}

// ---------------- Host launcher ----------------
extern "C" void kernel_launch(void* const* d_in, const int* in_sizes, int n_in,
                              void* d_out, int out_size, void* d_ws, size_t ws_size,
                              hipStream_t stream) {
  const float* x     = (const float*)d_in[0];
  const float* Wqkv  = (const float*)d_in[1];
  const float* bqkv  = (const float*)d_in[2];
  const float* Wproj = (const float*)d_in[3];
  const float* bproj = (const float*)d_in[4];

  float* out   = (float*)d_out;
  float* probs = out + (size_t)Bq * Sq * Dq;            // 4,194,304 floats in

  char* ws = (char*)d_ws;                                // 40 MB total layout:
  bf16* Wqkv_t  = (bf16*)(ws);                           //  6,291,456 B
  bf16* Wproj_t = (bf16*)(ws +  6291456);                //  2,097,152 B
  bf16* Q       = (bf16*)(ws +  8388608);                //  8,388,608 B
  bf16* K       = (bf16*)(ws + 16777216);                //  8,388,608 B
  bf16* Vt      = (bf16*)(ws + 25165824);                //  8,388,608 B
  bf16* ctxb    = (bf16*)(ws + 33554432);                //  8,388,608 B

  // 1) weight convert + transpose (bf16)
  cvt_transpose_k<<<(N3D * Dq) / 256, 256, 0, stream>>>(Wqkv,  Wqkv_t,  N3D);
  cvt_transpose_k<<<(Dq  * Dq) / 256, 256, 0, stream>>>(Wproj, Wproj_t, Dq);

  // 2) QKV GEMM -> Q[B,H,S,HD], K[B,H,S,HD], Vt[B,H,HD,S]
  gemm_qkv_k<<<dim3(N3D / 64, MROWS / 64), 128, 0, stream>>>(x, Wqkv_t, bqkv, Q, K, Vt);

  // 3) fused attention (scores -> softmax -> probs out -> ctx)
  const int smem_bytes = (16 * SROW + 4 * 1024) * (int)sizeof(float);  // 147,712 B
  hipFuncSetAttribute((const void*)attn_k,
                      hipFuncAttributeMaxDynamicSharedMemorySize, smem_bytes);
  attn_k<<<Bq * Hq * (Sq / 16), 128, smem_bytes, stream>>>(Q, K, Vt, probs, ctxb);

  // 4) output projection
  gemm_proj_k<<<dim3(Dq / 64, MROWS / 64), 128, 0, stream>>>(ctxb, Wproj_t, bproj, out);
}
